// GTModel_9079560864211
// MI455X (gfx1250) — matrix operand order repaired and checked by reading, not compiled, not run on hardware
//
#include <hip/hip_runtime.h>
#include <math.h>

// ---------------- problem constants (match reference) ----------------
#define N_NODES 10000
#define N_EDGES 160000
#define IN_DIM  128
#define HID     128
#define NHEAD   8
#define HDIM    16
#define NLAYER  8
#define PDIM    8
#define FFN     256
#define BN_EPS  1e-5f
#define QSCALE  0.25f   // HD^-0.5 = 16^-0.5

typedef __bf16 bf16_t;
typedef bf16_t v16bf __attribute__((ext_vector_type(16)));
typedef bf16_t v8bf  __attribute__((ext_vector_type(8)));
typedef float  v8f   __attribute__((ext_vector_type(8)));
typedef unsigned int u32x4 __attribute__((ext_vector_type(4)));
typedef int          i32x4 __attribute__((ext_vector_type(4)));
typedef int          i32x8 __attribute__((ext_vector_type(8)));

// ---------------------------------------------------------------
// WMMA GEMM:  C[N,M] = act( (A[N,K] @ B[K,M] + bias) * scale + resid )
//
// A:  bf16 row-major [N,K]  (streamed from global, L2-resident)
// Bt: bf16 row-major [M,K]  (B transposed; 128-column slab staged into LDS
//                            once per block via the Tensor Data Mover)
//
// Block = 64 threads (2 waves). Each wave owns 16 rows x 64 cols
// (4 WMMA accumulators; one A fragment feeds 4 v_wmma per K-step;
//  all 4 B fragments loaded up front so DS loads clause together and
//  the 4 WMMAs issue back-to-back).
// Each block loops over 5 row tiles reusing the LDS weight slab.
// grid = (N/80, M/128).  K mult of 32, N mult of 80.
// act: 0=none, 1=relu, 2=leaky(0.1)
// ---------------------------------------------------------------
__global__ void __launch_bounds__(64)
gemm_bf16_wmma(const bf16_t* __restrict__ A, const bf16_t* __restrict__ Bt,
               const float* __restrict__ bias, const float* __restrict__ resid,
               float* __restrict__ Cf, bf16_t* __restrict__ Cb,
               int K, int M, float scale, int act)
{
    extern __shared__ char smem_dyn[];          // 128 * K bf16 weight slab
    bf16_t* Bs = (bf16_t*)smem_dyn;

    const int wave = threadIdx.x >> 5;
    const int lane = threadIdx.x & 31;
    const int half = lane >> 4;                 // 0: lanes 0-15, 1: lanes 16-31
    const int r    = lane & 15;
    const int colBlock  = blockIdx.y << 7;      // 128 cols per block
    const int tileElems = K << 7;               // 128 * K (<= 32768, fits tile_dim0)

    // ---- stage Bt[colBlock .. colBlock+127][0..K-1] into LDS via TDM ----
    if (wave == 0) {
        unsigned long long ga =
            (unsigned long long)(const void*)(Bt + (size_t)colBlock * K);
        u32x4 g0 = {0u, 0u, 0u, 0u};
        g0[0] = 1u;                                             // count = 1
        g0[1] = 0u;                                             // lds_addr = 0 (dyn-LDS base)
        g0[2] = (unsigned)ga;                                   // global_addr[31:0]
        g0[3] = (unsigned)((ga >> 32) & 0x1FFFFFFull) | (2u << 30); // addr[56:32] | type=2
        i32x8 g1 = {0, 0, 0, 0, 0, 0, 0, 0};
        g1[0] = (int)(1u << 16);                                // data_size = 2 bytes
        g1[1] = (int)(((unsigned)tileElems & 0xFFFFu) << 16);   // tensor_dim0[15:0]
        g1[2] = (int)((((unsigned)tileElems >> 16) & 0xFFFFu) | (1u << 16)); // dim0 hi | tensor_dim1=1
        g1[3] = (int)(((unsigned)tileElems & 0xFFFFu) << 16);   // tile_dim0
        g1[4] = 1;                                              // tile_dim1 = 1
        g1[5] = tileElems;                                      // tensor_dim0_stride
        i32x4 g2 = {0, 0, 0, 0};
        i32x4 g3 = {0, 0, 0, 0};
#if __clang_major__ >= 23
        i32x8 g4 = {0, 0, 0, 0, 0, 0, 0, 0};
        __builtin_amdgcn_tensor_load_to_lds(g0, g1, g2, g3, g4, 0);
#else
        __builtin_amdgcn_tensor_load_to_lds(g0, g1, g2, g3, 0);
#endif
        __builtin_amdgcn_s_wait_tensorcnt(0);
    }
    __syncthreads();

    // B fragment base for this wave (column = wave*64 + j*16 + r)
    const bf16_t* Bsw = Bs + (size_t)((wave << 6) + r) * K + (half << 3);

    for (int t = 0; t < 5; ++t) {
        const int row0 = (blockIdx.x * 5 + t) << 4;
        const bf16_t* pa = A + (size_t)(row0 + r) * K + (half << 3);

        const v8f zero = {0.f, 0.f, 0.f, 0.f, 0.f, 0.f, 0.f, 0.f};
        v8f acc[4];
#pragma unroll
        for (int j = 0; j < 4; ++j) acc[j] = zero;

        for (int k = 0; k < K; k += 32) {
            // A fragment: lane<16 K{0..7,16..23}, lane>=16 K{8..15,24..31}
            v8bf a0 = *(const v8bf*)(pa + k);
            v8bf a1 = *(const v8bf*)(pa + k + 16);
            v16bf af;
#pragma unroll
            for (int i = 0; i < 8; ++i) { af[i] = a0[i]; af[i + 8] = a1[i]; }

            // all 4 B fragments up front (distinct registers -> one DS clause)
            v16bf bv[4];
#pragma unroll
            for (int j = 0; j < 4; ++j) {
                // B fragment: lane<16 K{0..7,16..23}, lane>=16 K{8..15,24..31} of its column
                const bf16_t* pb = Bsw + (size_t)(j << 4) * K + k;
                v8bf b0 = *(const v8bf*)pb;
                v8bf b1 = *(const v8bf*)(pb + 16);
#pragma unroll
                for (int i = 0; i < 8; ++i) { bv[j][i] = b0[i]; bv[j][i + 8] = b1[i]; }
            }

            // 4 back-to-back WMMAs sharing the A fragment
#pragma unroll
            for (int j = 0; j < 4; ++j) {
                acc[j] = __builtin_amdgcn_wmma_f32_16x16x32_bf16(
                    false, af, false, bv[j], (short)0, acc[j], false, false);
            }
        }

        // C layout: acc[j][v] -> row = row0 + half*8 + v, col = colBlock + wave*64 + j*16 + r
#pragma unroll
        for (int j = 0; j < 4; ++j) {
            const int cc = colBlock + (wave << 6) + (j << 4) + r;
            const float bval = bias ? bias[cc] : 0.f;
#pragma unroll
            for (int v = 0; v < 8; ++v) {
                const int rr = row0 + (half << 3) + v;
                const size_t idx = (size_t)rr * M + cc;
                float val = (acc[j][v] + bval) * scale;
                if (resid) val += resid[idx];
                if (act == 1)      val = fmaxf(val, 0.f);
                else if (act == 2) val = (val > 0.f) ? val : 0.1f * val;
                if (Cf) Cf[idx] = val;
                if (Cb) Cb[idx] = (bf16_t)val;
            }
        }
    }
}

// ---------------- weight transpose + f32->bf16 ----------------
__global__ void convW_kernel(const float* __restrict__ W, bf16_t* __restrict__ Wt,
                             int K, int M, float scale)
{
    int t = blockIdx.x * blockDim.x + threadIdx.x;
    if (t >= K * M) return;
    int m = t / K, k = t - m * K;               // output index [m][k]
    Wt[t] = (bf16_t)(W[(size_t)k * M + m] * scale);
}

// ---------------- generic f32 -> bf16 ----------------
__global__ void conv_bf_kernel(const float* __restrict__ x, bf16_t* __restrict__ y, int n)
{
    int t = blockIdx.x * blockDim.x + threadIdx.x;
    if (t < n) y[t] = (bf16_t)x[t];
}

// ---------------- h += pos_enc @ pos_W + pos_b ; also emit bf16 ----------------
__global__ void add_pos_kernel(float* __restrict__ h, const float* __restrict__ pe,
                               const float* __restrict__ pW, const float* __restrict__ pb,
                               bf16_t* __restrict__ hbf)
{
    int t = blockIdx.x * blockDim.x + threadIdx.x;
    if (t >= N_NODES * HID) return;
    int n = t >> 7, c = t & 127;
    float a = pb[c];
    const float* pen = pe + n * PDIM;
#pragma unroll
    for (int p = 0; p < PDIM; ++p) a += pen[p] * pW[p * HID + c];
    float val = h[t] + a;
    h[t] = val;
    hbf[t] = (bf16_t)val;
}

// ---------------- attention scratch init ----------------
__global__ void init_attn_kernel(float* __restrict__ maxb, float* __restrict__ den,
                                 float* __restrict__ o)
{
    int t = blockIdx.x * blockDim.x + threadIdx.x;
    if (t < N_NODES * NHEAD) { maxb[t] = -3.0e38f; den[t] = 0.f; }
    if (t < N_NODES * HID)   o[t] = 0.f;
}

__device__ inline void atomicMaxF(float* addr, float val)
{
    unsigned int* ua = (unsigned int*)addr;
    unsigned int old = *ua;
    while (true) {
        float f = __uint_as_float(old);
        if (f >= val) break;
        unsigned int prev = atomicCAS(ua, old, __float_as_uint(val));
        if (prev == old) break;
        old = prev;
    }
}

// s[e,h] = sum_d q[src][d*8+h] * k[dst][d*8+h] ; atomic max per (src,h)
__global__ void edge_score_kernel(const float* __restrict__ q, const float* __restrict__ k,
                                  const int* __restrict__ src, const int* __restrict__ dst,
                                  float* __restrict__ s, float* __restrict__ maxb)
{
    int t = blockIdx.x * blockDim.x + threadIdx.x;
    if (t >= N_EDGES * NHEAD) return;
    int e = t >> 3, hh = t & 7;
    int a = src[e], b = dst[e];
    const float* qp = q + (size_t)a * HID + hh;
    const float* kp = k + (size_t)b * HID + hh;
    float acc = 0.f;
#pragma unroll
    for (int d = 0; d < HDIM; ++d) acc += qp[d * NHEAD] * kp[d * NHEAD];
    s[t] = acc;
    atomicMaxF(&maxb[a * NHEAD + hh], acc);
}

// ex = exp(s - max[src]) ; atomic add denominator
__global__ void edge_exp_kernel(const float* __restrict__ s, const float* __restrict__ maxb,
                                const int* __restrict__ src,
                                float* __restrict__ ex, float* __restrict__ den)
{
    int t = blockIdx.x * blockDim.x + threadIdx.x;
    if (t >= N_EDGES * NHEAD) return;
    int e = t >> 3, hh = t & 7;
    int a = src[e];
    float v = __expf(s[t] - maxb[a * NHEAD + hh]);
    ex[t] = v;
    atomicAdd(&den[a * NHEAD + hh], v);
}

// o[src, d, h] += (ex/den[src]) * v[dst, d, h]
__global__ void edge_agg_kernel(const float* __restrict__ ex, const float* __restrict__ den,
                                const int* __restrict__ src, const int* __restrict__ dst,
                                const float* __restrict__ v, float* __restrict__ o)
{
    int t = blockIdx.x * blockDim.x + threadIdx.x;
    if (t >= N_EDGES * NHEAD) return;
    int e = t >> 3, hh = t & 7;
    int a = src[e], b = dst[e];
    float attn = ex[t] / den[a * NHEAD + hh];
    const float* vp = v + (size_t)b * HID + hh;
    float* op = o + (size_t)a * HID + hh;
#pragma unroll
    for (int d = 0; d < HDIM; ++d) atomicAdd(&op[d * NHEAD], attn * vp[d * NHEAD]);
}

// ---------------- per-column batch stats (biased variance) ----------------
__global__ void bn_stats_kernel(const float* __restrict__ x, float* __restrict__ mean,
                                float* __restrict__ var)
{
    __shared__ float sh[256], sh2[256];
    int c = blockIdx.x;
    float s = 0.f, s2 = 0.f;
    for (int n = threadIdx.x; n < N_NODES; n += 256) {
        float v = x[(size_t)n * HID + c];
        s += v; s2 += v * v;
    }
    sh[threadIdx.x] = s; sh2[threadIdx.x] = s2;
    __syncthreads();
    for (int off = 128; off > 0; off >>= 1) {
        if (threadIdx.x < off) {
            sh[threadIdx.x]  += sh[threadIdx.x + off];
            sh2[threadIdx.x] += sh2[threadIdx.x + off];
        }
        __syncthreads();
    }
    if (threadIdx.x == 0) {
        float m = sh[0] * (1.f / N_NODES);
        mean[c] = m;
        var[c]  = sh2[0] * (1.f / N_NODES) - m * m;
    }
}

__global__ void bn_apply_kernel(const float* __restrict__ x, const float* __restrict__ mean,
                                const float* __restrict__ var, const float* __restrict__ g,
                                const float* __restrict__ b, float* __restrict__ h,
                                bf16_t* __restrict__ hbf)
{
    int t = blockIdx.x * blockDim.x + threadIdx.x;
    if (t >= N_NODES * HID) return;
    int c = t & 127;
    float iv = rsqrtf(var[c] + BN_EPS);
    float val = g[c] * (x[t] - mean[c]) * iv + b[c];
    h[t] = val;
    hbf[t] = (bf16_t)val;
}

// ---------------- column sum pooling ----------------
__global__ void colsum_kernel(const float* __restrict__ x, float* __restrict__ pooled)
{
    __shared__ float sh[256];
    int c = blockIdx.x;
    float s = 0.f;
    for (int n = threadIdx.x; n < N_NODES; n += 256) s += x[(size_t)n * HID + c];
    sh[threadIdx.x] = s;
    __syncthreads();
    for (int off = 128; off > 0; off >>= 1) {
        if (threadIdx.x < off) sh[threadIdx.x] += sh[threadIdx.x + off];
        __syncthreads();
    }
    if (threadIdx.x == 0) pooled[c] = sh[0];
}

// ---------------- predictor head: 128 -> 64 -> 32 -> 1 ----------------
__global__ void predictor_kernel(const float* __restrict__ pooled,
                                 const float* __restrict__ W1, const float* __restrict__ b1,
                                 const float* __restrict__ W2, const float* __restrict__ b2,
                                 const float* __restrict__ W3, const float* __restrict__ b3,
                                 float* __restrict__ out)
{
    __shared__ float x1[64], x2[32];
    int t = threadIdx.x;
    if (t < 64) {
        float a = b1[t];
        for (int i = 0; i < 128; ++i) a += pooled[i] * W1[i * 64 + t];
        x1[t] = fmaxf(a, 0.f);
    }
    __syncthreads();
    if (t < 32) {
        float a = b2[t];
        for (int i = 0; i < 64; ++i) a += x1[i] * W2[i * 32 + t];
        x2[t] = fmaxf(a, 0.f);
    }
    __syncthreads();
    if (t == 0) {
        float a = b3[0];
        for (int i = 0; i < 32; ++i) a += x2[i] * W3[i];
        out[0] = a;
    }
}

// =====================================================================
extern "C" void kernel_launch(void* const* d_in, const int* in_sizes, int n_in,
                              void* d_out, int out_size, void* d_ws, size_t ws_size,
                              hipStream_t stream)
{
    (void)in_sizes; (void)n_in; (void)out_size; (void)ws_size;

    const float* X       = (const float*)d_in[0];
    const float* pos_enc = (const float*)d_in[1];
    const int*   src     = (const int*)d_in[2];
    const int*   dst     = (const int*)d_in[3];
    const float* enc_W1  = (const float*)d_in[4];
    const float* enc_b1  = (const float*)d_in[5];
    const float* enc_W2  = (const float*)d_in[6];
    const float* enc_b2  = (const float*)d_in[7];
    const float* pos_W   = (const float*)d_in[8];
    const float* pos_b   = (const float*)d_in[9];
    const float* Wq      = (const float*)d_in[10];
    const float* bq      = (const float*)d_in[11];
    const float* Wk      = (const float*)d_in[12];
    const float* bk      = (const float*)d_in[13];
    const float* Wv      = (const float*)d_in[14];
    const float* bv      = (const float*)d_in[15];
    const float* Wo      = (const float*)d_in[16];
    const float* bo      = (const float*)d_in[17];
    const float* bn1_g   = (const float*)d_in[18];
    const float* bn1_b   = (const float*)d_in[19];
    const float* bn2_g   = (const float*)d_in[20];
    const float* bn2_b   = (const float*)d_in[21];
    const float* W_ffn1  = (const float*)d_in[22];
    const float* b_ffn1  = (const float*)d_in[23];
    const float* W_ffn2  = (const float*)d_in[24];
    const float* b_ffn2  = (const float*)d_in[25];
    const float* pred_W1 = (const float*)d_in[26];
    const float* pred_b1 = (const float*)d_in[27];
    const float* pred_W2 = (const float*)d_in[28];
    const float* pred_b2 = (const float*)d_in[29];
    const float* pred_W3 = (const float*)d_in[30];
    const float* pred_b3 = (const float*)d_in[31];
    float* out = (float*)d_out;

    // ---------- deterministic bump allocator over d_ws ----------
    char* wsp = (char*)d_ws;
    auto alloc = [&](size_t bytes) -> void* {
        void* p = (void*)wsp;
        wsp += (bytes + 255) & ~(size_t)255;
        return p;
    };

    const size_t NH128 = (size_t)N_NODES * HID;
    const size_t NH256 = (size_t)N_NODES * FFN;
    const size_t E8    = (size_t)N_EDGES * NHEAD;

    bf16_t* Xbf    = (bf16_t*)alloc(NH128 * 2);
    bf16_t* hbf    = (bf16_t*)alloc(NH128 * 2);
    bf16_t* obf    = (bf16_t*)alloc(NH128 * 2);   // also encoder hidden
    bf16_t* tbf    = (bf16_t*)alloc(NH256 * 2);
    bf16_t* encW1t = (bf16_t*)alloc(128 * 128 * 2);
    bf16_t* encW2t = (bf16_t*)alloc(128 * 128 * 2);
    bf16_t* Wqt    = (bf16_t*)alloc((size_t)NLAYER * 128 * 128 * 2);
    bf16_t* Wkt    = (bf16_t*)alloc((size_t)NLAYER * 128 * 128 * 2);
    bf16_t* Wvt    = (bf16_t*)alloc((size_t)NLAYER * 128 * 128 * 2);
    bf16_t* Wot    = (bf16_t*)alloc((size_t)NLAYER * 128 * 128 * 2);
    bf16_t* Wf1t   = (bf16_t*)alloc((size_t)NLAYER * 128 * 256 * 2);
    bf16_t* Wf2t   = (bf16_t*)alloc((size_t)NLAYER * 256 * 128 * 2);

    float* h    = (float*)alloc(NH128 * 4);
    float* qb   = (float*)alloc(NH128 * 4);
    float* kb   = (float*)alloc(NH128 * 4);
    float* vb   = (float*)alloc(NH128 * 4);
    float* ob   = (float*)alloc(NH128 * 4);
    float* pre  = (float*)alloc(NH128 * 4);
    float* sb   = (float*)alloc(E8 * 4);
    float* exb  = (float*)alloc(E8 * 4);
    float* maxb = (float*)alloc((size_t)N_NODES * NHEAD * 4);
    float* den  = (float*)alloc((size_t)N_NODES * NHEAD * 4);
    float* mean = (float*)alloc(128 * 4);
    float* var  = (float*)alloc(128 * 4);
    float* pooled = (float*)alloc(128 * 4);

    const dim3 blk(256);
    const dim3 gNH128((unsigned)((NH128 + 255) / 256));
    const dim3 gE8((unsigned)((E8 + 255) / 256));

    // GEMM launch helper: grid=(N/80, M/128), block=64, dyn-LDS = 128*K*2
    auto gemm = [&](const bf16_t* A, const bf16_t* Bt, const float* bias,
                    const float* resid, float* Cf, bf16_t* Cb,
                    int K, int M, float scale, int act) {
        dim3 grid(N_NODES / 80, M / 128);
        size_t sh = (size_t)128 * K * 2;
        gemm_bf16_wmma<<<grid, 64, sh, stream>>>(A, Bt, bias, resid, Cf, Cb,
                                                 K, M, scale, act);
    };

    auto convw = [&](const float* W, bf16_t* Wt, int K, int M) {
        int total = K * M;
        convW_kernel<<<(total + 255) / 256, blk, 0, stream>>>(W, Wt, K, M, 1.f);
    };

    // ---------- weight conversion (every call; deterministic) ----------
    convw(enc_W1, encW1t, 128, 128);
    convw(enc_W2, encW2t, 128, 128);
    for (int l = 0; l < NLAYER; ++l) {
        size_t s128 = (size_t)l * 128 * 128;
        size_t s256 = (size_t)l * 128 * 256;
        convw(Wq + s128, Wqt + s128, 128, 128);
        convw(Wk + s128, Wkt + s128, 128, 128);
        convw(Wv + s128, Wvt + s128, 128, 128);
        convw(Wo + s128, Wot + s128, 128, 128);
        convw(W_ffn1 + s256, Wf1t + s256, 128, 256);
        convw(W_ffn2 + s256, Wf2t + s256, 256, 128);
    }

    // ---------- encoder ----------
    conv_bf_kernel<<<gNH128, blk, 0, stream>>>(X, Xbf, (int)NH128);
    // t1 = leaky_relu(X @ W1 + b1), bf16 only
    gemm(Xbf, encW1t, enc_b1, nullptr, nullptr, obf, 128, 128, 1.f, 2);
    // h = t1 @ W2 + b2
    gemm(obf, encW2t, enc_b2, nullptr, h, nullptr, 128, 128, 1.f, 0);
    // h += pos_enc @ pos_W + pos_b (also hbf)
    add_pos_kernel<<<gNH128, blk, 0, stream>>>(h, pos_enc, pos_W, pos_b, hbf);

    // ---------- transformer layers ----------
    for (int l = 0; l < NLAYER; ++l) {
        size_t s128 = (size_t)l * 128 * 128;
        size_t s256 = (size_t)l * 128 * 256;
        const float* bql = bq + (size_t)l * 128;
        const float* bkl = bk + (size_t)l * 128;
        const float* bvl = bv + (size_t)l * 128;
        const float* bol = bo + (size_t)l * 128;

        // q/k/v projections (q folded with 1/sqrt(HD))
        gemm(hbf, Wqt + s128, bql, nullptr, qb, nullptr, 128, 128, QSCALE, 0);
        gemm(hbf, Wkt + s128, bkl, nullptr, kb, nullptr, 128, 128, 1.f, 0);
        gemm(hbf, Wvt + s128, bvl, nullptr, vb, nullptr, 128, 128, 1.f, 0);

        // sparse attention: score -> segment softmax -> aggregate
        init_attn_kernel<<<gNH128, blk, 0, stream>>>(maxb, den, ob);
        edge_score_kernel<<<gE8, blk, 0, stream>>>(qb, kb, src, dst, sb, maxb);
        edge_exp_kernel<<<gE8, blk, 0, stream>>>(sb, maxb, src, exb, den);
        edge_agg_kernel<<<gE8, blk, 0, stream>>>(exb, den, src, dst, vb, ob);

        // output projection + residual, then BN1
        conv_bf_kernel<<<gNH128, blk, 0, stream>>>(ob, obf, (int)NH128);
        gemm(obf, Wot + s128, bol, h, pre, nullptr, 128, 128, 1.f, 0);
        bn_stats_kernel<<<128, blk, 0, stream>>>(pre, mean, var);
        bn_apply_kernel<<<gNH128, blk, 0, stream>>>(pre, mean, var,
                                                    bn1_g + (size_t)l * 128,
                                                    bn1_b + (size_t)l * 128, h, hbf);

        // FFN: relu(h @ W1 + b1) @ W2 + b2, residual, BN2
        gemm(hbf, Wf1t + s256, b_ffn1 + (size_t)l * 256, nullptr,
             nullptr, tbf, 128, 256, 1.f, 1);
        gemm(tbf, Wf2t + s256, b_ffn2 + (size_t)l * 128, h,
             pre, nullptr, 256, 128, 1.f, 0);
        bn_stats_kernel<<<128, blk, 0, stream>>>(pre, mean, var);
        bn_apply_kernel<<<gNH128, blk, 0, stream>>>(pre, mean, var,
                                                    bn2_g + (size_t)l * 128,
                                                    bn2_b + (size_t)l * 128, h, hbf);
    }

    // ---------- pooling + predictor ----------
    colsum_kernel<<<128, blk, 0, stream>>>(h, pooled);
    predictor_kernel<<<1, 64, 0, stream>>>(pooled, pred_W1, pred_b1,
                                           pred_W2, pred_b2, pred_W3, pred_b3, out);
}
